// GATNet_85890755986006
// MI455X (gfx1250) — matrix operand order repaired and checked by reading, not compiled, not run on hardware
//
#include <hip/hip_runtime.h>
#include <hip/hip_bf16.h>
#include <cstdint>

#define DIMV     256
#define NHEADSV  2
#define NLAYERSV 2
#define ALPHAV   0.2f
#define MT       32          // rows of X staged per workgroup by the TDM

typedef __attribute__((ext_vector_type(16))) _Float16 v16h;
typedef __attribute__((ext_vector_type(8)))  float    v8f;
typedef __attribute__((ext_vector_type(4)))  unsigned int u32x4;
typedef __attribute__((ext_vector_type(8)))  int      i32x8;
typedef __attribute__((ext_vector_type(4)))  int      i32x4;

// ---------------------------------------------------------------------------
// utility kernels
// ---------------------------------------------------------------------------
__global__ void fill_kernel(float* __restrict__ p, float v, int n) {
    int i = blockIdx.x * blockDim.x + threadIdx.x;
    if (i < n) p[i] = v;
}

__global__ void f32_to_f16_kernel(const float* __restrict__ in,
                                  _Float16* __restrict__ out, int n) {
    int i = blockIdx.x * blockDim.x + threadIdx.x;
    if (i < n) out[i] = (_Float16)in[i];
}

// W[l] layout: [H, D, D] row-major over (k, n). Produce WT: [H, D, D] over (n, k)
// so B-fragment pair loads (k, k+1) are contiguous.
__global__ void conv_w_kernel(const float* __restrict__ W,
                              _Float16* __restrict__ WT) {
    int i = blockIdx.x * blockDim.x + threadIdx.x;     // over H*D*D
    if (i < NHEADSV * DIMV * DIMV) {
        int h = i >> 16;
        int k = (i >> 8) & 255;
        int n = i & 255;
        WT[(h << 16) | (n << 8) | k] = (_Float16)W[i];
    }
}

// ---------------------------------------------------------------------------
// WMMA GEMM:  Hout[head] = X(f16)[M,256] @ W[head](f16)[256,256], f32 accum
// grid = (ceil(M/32), NHEADS), block = 256 (8 waves).
// A-tile (32x256 f16) staged into LDS by the Tensor Data Mover; each wave
// computes a 32x32 output tile (4 accumulators), B fragments from L2-hot W.
// ---------------------------------------------------------------------------
__global__ __launch_bounds__(256) void gemm_wmma_kernel(
    const _Float16* __restrict__ X,    // [M, 256]
    const _Float16* __restrict__ WT,   // [H, 256(n), 256(k)]
    float* __restrict__ Hout,          // [H, M, 256]
    int M)
{
    __shared__ _Float16 Atile[MT * DIMV];      // 16 KB, LDS offset 0

    const int head = blockIdx.y;
    const int m0   = blockIdx.x * MT;
    const int wave = threadIdx.x >> 5;
    const int lane = threadIdx.x & 31;
    const int lm   = lane & 15;
    const int g    = lane >> 4;
    const int n0   = wave << 5;

    if (wave == 0) {
        // --- Tensor DMA descriptor (ISA 8.3/8.4): 2-D tile 256 x MT of 2-byte
        // elements from row-major X starting at row m0; OOB rows zero-fill.
        unsigned long long ga = (unsigned long long)(const void*)(X + (size_t)m0 * DIMV);
        u32x4 g0;
        g0[0] = 1u;                                    // count=1 (valid user D#)
        g0[1] = 0u;                                    // lds_addr = 0 (Atile at base)
        g0[2] = (unsigned)(ga & 0xFFFFFFFFu);          // global_addr[31:0]
        g0[3] = (unsigned)((ga >> 32) & 0x01FFFFFFu)   // global_addr[56:32]
              | (2u << 30);                            // type = 2 ("image")
        i32x8 g1;
        g1[0] = 1 << 16;                               // data_size=1 -> 2 bytes
        g1[1] = (DIMV & 0xFFFF) << 16;                 // tensor_dim0[15:0] = 256
        g1[2] = (DIMV >> 16) | ((M & 0xFFFF) << 16);   // dim0 hi | tensor_dim1 lo
        g1[3] = (M >> 16) | ((DIMV & 0xFFFF) << 16);   // dim1 hi | tile_dim0 = 256
        g1[4] = MT;                                    // tile_dim1 = 32, tile_dim2 = 0
        g1[5] = DIMV;                                  // tensor_dim0_stride = 256
        g1[6] = 0;
        g1[7] = 0;                                     // tensor_dim1_stride unused (2D)
        i32x4 z4 = {0, 0, 0, 0};                       // groups 2/3 unused (2-D tensor)
        i32x8 z8 = {0, 0, 0, 0, 0, 0, 0, 0};
        __builtin_amdgcn_tensor_load_to_lds(g0, g1, z4, z4, z8, 0);
        __builtin_amdgcn_s_wait_tensorcnt(0);
    }
    __syncthreads();
    // The TDM wrote Atile through a hardware path invisible to alias analysis
    // (the LDS address is a descriptor integer). Escape the pointer + memory
    // clobber so the compiler treats Atile as written and emits real ds loads.
    asm volatile("" : : "v"((void*)Atile) : "memory");

    const _Float16* A0 = Atile + (size_t)lm * DIMV;          // rows 0..15 of tile
    const _Float16* A1 = Atile + (size_t)(16 + lm) * DIMV;   // rows 16..31
    const _Float16* W0 = WT + (((size_t)head << 16) | ((size_t)(n0 + lm) << 8));
    const _Float16* W1 = W0 + (16 << 8);

    v8f c00 = {}, c01 = {}, c10 = {}, c11 = {};
    for (int k0 = 0; k0 < DIMV; k0 += 32) {
        __builtin_prefetch(W0 + k0 + 32);
        v16h a0, a1, b0, b1;
#pragma unroll
        for (int i = 0; i < 8; ++i) {
            // A-matrix 16x32 f16 layout (ISA 7.12.2): group g holds K 8g..8g+7, 16+8g..
            int ka = k0 + ((i < 4) ? (2 * i) : (16 + 2 * (i - 4))) + (g << 3);
            a0[2 * i]     = A0[ka];
            a0[2 * i + 1] = A0[ka + 1];
            a1[2 * i]     = A1[ka];
            a1[2 * i + 1] = A1[ka + 1];
            // B-matrix 32x16 f16 layout: same K order as A (group g holds K 8g..8g+7, 16+8g..)
            int kb = k0 + ((i < 4) ? (2 * i) : (16 + 2 * (i - 4))) + (g << 3);
            b0[2 * i]     = W0[kb];
            b0[2 * i + 1] = W0[kb + 1];
            b1[2 * i]     = W1[kb];
            b1[2 * i + 1] = W1[kb + 1];
        }
        c00 = __builtin_amdgcn_wmma_f32_16x16x32_f16(false, a0, false, b0,
                                                     (short)0, c00, false, false);
        c01 = __builtin_amdgcn_wmma_f32_16x16x32_f16(false, a0, false, b1,
                                                     (short)0, c01, false, false);
        c10 = __builtin_amdgcn_wmma_f32_16x16x32_f16(false, a1, false, b0,
                                                     (short)0, c10, false, false);
        c11 = __builtin_amdgcn_wmma_f32_16x16x32_f16(false, a1, false, b1,
                                                     (short)0, c11, false, false);
    }

    float* Ho = Hout + (size_t)head * M * DIMV;
#pragma unroll
    for (int r = 0; r < 8; ++r) {
        int ma = m0 + r + (g << 3);
        int mb = ma + 16;
        if (ma < M) {
            Ho[(size_t)ma * DIMV + n0 + lm]      = c00[r];
            Ho[(size_t)ma * DIMV + n0 + 16 + lm] = c01[r];
        }
        if (mb < M) {
            Ho[(size_t)mb * DIMV + n0 + lm]      = c10[r];
            Ho[(size_t)mb * DIMV + n0 + 16 + lm] = c11[r];
        }
    }
}

// ---------------------------------------------------------------------------
// per-(head,node) attention logits: s_src = h.a_src, s_dst = h.a_dst
// one wave32 per (head,node); lane handles 8 contiguous dims
// ---------------------------------------------------------------------------
__global__ __launch_bounds__(256) void score_kernel(
    const float* __restrict__ H,                      // [NH, N, 256]
    const float* __restrict__ a_src,                  // [NH, 256]
    const float* __restrict__ a_dst,
    float* __restrict__ ssrc, float* __restrict__ sdst, int N)
{
    int wid  = blockIdx.x * (blockDim.x >> 5) + (threadIdx.x >> 5);
    int lane = threadIdx.x & 31;
    if (wid >= NHEADSV * N) return;
    int head = wid / N;
    int n    = wid - head * N;
    const float* hp = H + ((size_t)head * N + n) * DIMV + lane * 8;
    const float* as = a_src + head * DIMV + lane * 8;
    const float* ad = a_dst + head * DIMV + lane * 8;
    float s1 = 0.f, s2 = 0.f;
#pragma unroll
    for (int j = 0; j < 8; ++j) { float v = hp[j]; s1 += v * as[j]; s2 += v * ad[j]; }
#pragma unroll
    for (int off = 16; off > 0; off >>= 1) {
        s1 += __shfl_xor(s1, off, 32);
        s2 += __shfl_xor(s2, off, 32);
    }
    if (lane == 0) { ssrc[wid] = s1; sdst[wid] = s2; }
}

// ---------------------------------------------------------------------------
// edge softmax (stable): max, then exp/sum, then weighted aggregation
// ---------------------------------------------------------------------------
__device__ __forceinline__ float lrelu(float x) { return x > 0.f ? x : ALPHAV * x; }

__device__ __forceinline__ void atomicMaxF(float* addr, float val) {
    int* ia = (int*)addr;
    int old = __float_as_int(*addr);
    while (__int_as_float(old) < val) {
        int assumed = old;
        old = atomicCAS(ia, assumed, __float_as_int(val));
        if (old == assumed) break;
    }
}

__global__ void edge_max_kernel(const int* __restrict__ src, const int* __restrict__ dst,
                                const float* __restrict__ ssrc, const float* __restrict__ sdst,
                                float* __restrict__ mbuf, int E, int N) {
    int e = blockIdx.x * blockDim.x + threadIdx.x;
    if (e >= E) return;
    int s = src[e], d = dst[e];
#pragma unroll
    for (int h = 0; h < NHEADSV; ++h) {
        float el = lrelu(ssrc[h * N + s] + sdst[h * N + d]);
        atomicMaxF(&mbuf[h * N + d], el);
    }
}

__global__ void edge_exp_kernel(const int* __restrict__ src, const int* __restrict__ dst,
                                const float* __restrict__ ssrc, const float* __restrict__ sdst,
                                const float* __restrict__ mbuf,
                                float* __restrict__ exbuf, float* __restrict__ den,
                                int E, int N) {
    int e = blockIdx.x * blockDim.x + threadIdx.x;
    if (e >= E) return;
    int s = src[e], d = dst[e];
#pragma unroll
    for (int h = 0; h < NHEADSV; ++h) {
        float el = lrelu(ssrc[h * N + s] + sdst[h * N + d]);
        float ex = __expf(el - mbuf[h * N + d]);
        exbuf[(size_t)h * E + e] = ex;
        atomicAdd(&den[h * N + d], ex);
    }
}

// block = 256 threads = 4 groups of 64; each group handles one (edge,head),
// each thread handles 4 contiguous dims (float4 gather + 4 atomic adds)
__global__ __launch_bounds__(256) void edge_aggr_kernel(
    const int* __restrict__ src, const int* __restrict__ dst,
    const float* __restrict__ exbuf, const float* __restrict__ den,
    const float* __restrict__ H, float* __restrict__ outacc, int E, int N)
{
    int task = blockIdx.x * 4 + (threadIdx.x >> 6);
    if (task >= NHEADSV * E) return;
    int h = task & 1;
    int e = task >> 1;
    int t = threadIdx.x & 63;
    int s = src[e], d = dst[e];
    float attn = exbuf[(size_t)h * E + e] / (den[h * N + d] + 1e-16f);
    const float4 hv = *(const float4*)(H + ((size_t)h * N + s) * DIMV + t * 4);
    float* op = outacc + (size_t)d * DIMV + t * 4;
    atomicAdd(op + 0, hv.x * attn);
    atomicAdd(op + 1, hv.y * attn);
    atomicAdd(op + 2, hv.z * attn);
    atomicAdd(op + 3, hv.w * attn);
}

// x = elu(mean over heads) = elu(0.5 * accumulated)
__global__ void elu_mean_kernel(const float* __restrict__ outacc,
                                float* __restrict__ xf, int n) {
    int i = blockIdx.x * blockDim.x + threadIdx.x;
    if (i < n) {
        float v = 0.5f * outacc[i];
        xf[i] = v > 0.f ? v : expm1f(v);
    }
}

// ---------------------------------------------------------------------------
// outputs
// ---------------------------------------------------------------------------
__global__ void gather_kernel(const float* __restrict__ x, const int* __restrict__ idx,
                              float* __restrict__ out, int total) {
    int i = blockIdx.x * blockDim.x + threadIdx.x;
    if (i >= total) return;
    int row = i >> 8, d = i & 255;
    out[i] = x[(size_t)idx[row] * DIMV + d];
}

__global__ void transe_kernel(const float* __restrict__ ent, const float* __restrict__ rel,
                              const int* __restrict__ hh, const int* __restrict__ tt,
                              const int* __restrict__ rr,
                              float* __restrict__ out, int total) {
    int i = blockIdx.x * blockDim.x + threadIdx.x;
    if (i >= total) return;
    int tri = i >> 8, d = i & 255;
    out[i] = ent[(size_t)hh[tri] * DIMV + d] + rel[(size_t)rr[tri] * DIMV + d]
           - ent[(size_t)tt[tri] * DIMV + d];
}

// ---------------------------------------------------------------------------
// host orchestration
// ---------------------------------------------------------------------------
extern "C" void kernel_launch(void* const* d_in, const int* in_sizes, int n_in,
                              void* d_out, int out_size, void* d_ws, size_t ws_size,
                              hipStream_t stream) {
    const float* emb[2]  = {(const float*)d_in[0], (const float*)d_in[1]};
    const float* rel[2]  = {(const float*)d_in[2], (const float*)d_in[3]};
    const float* Wall    = (const float*)d_in[4];
    const float* a_src   = (const float*)d_in[5];
    const float* a_dst   = (const float*)d_in[6];
    const int* srcs[2]   = {(const int*)d_in[7], (const int*)d_in[9]};
    const int* dsts[2]   = {(const int*)d_in[8], (const int*)d_in[10]};
    const int* datas[2]  = {(const int*)d_in[11], (const int*)d_in[12]};
    const int* hhs[2]    = {(const int*)d_in[13], (const int*)d_in[16]};
    const int* tts[2]    = {(const int*)d_in[14], (const int*)d_in[17]};
    const int* rrs[2]    = {(const int*)d_in[15], (const int*)d_in[18]};

    const int Ns[2]   = {in_sizes[0] / DIMV, in_sizes[1] / DIMV};
    const int Es[2]   = {in_sizes[7], in_sizes[9]};
    const int NB[2]   = {in_sizes[11], in_sizes[12]};
    const int NTRI[2] = {in_sizes[13], in_sizes[16]};

    const int Nmax = Ns[0] > Ns[1] ? Ns[0] : Ns[1];
    const int Emax = Es[0] > Es[1] ? Es[0] : Es[1];

    // workspace carve (256B aligned)
    char* w = (char*)d_ws;
    auto carve = [&](size_t bytes) -> void* {
        void* p = (void*)w;
        w += (bytes + 255) & ~(size_t)255;
        return p;
    };
    float*    xf     = (float*)   carve((size_t)Nmax * DIMV * 4);
    _Float16* xh     = (_Float16*)carve((size_t)Nmax * DIMV * 2);
    _Float16* WT     = (_Float16*)carve((size_t)NHEADSV * DIMV * DIMV * 2);
    float*    Hbuf   = (float*)   carve((size_t)NHEADSV * Nmax * DIMV * 4);
    float*    ssrc   = (float*)   carve((size_t)NHEADSV * Nmax * 4);
    float*    sdst   = (float*)   carve((size_t)NHEADSV * Nmax * 4);
    float*    mbuf   = (float*)   carve((size_t)NHEADSV * Nmax * 4);
    float*    den    = (float*)   carve((size_t)NHEADSV * Nmax * 4);
    float*    exbuf  = (float*)   carve((size_t)NHEADSV * Emax * 4);
    float*    outacc = (float*)   carve((size_t)Nmax * DIMV * 4);

    float* outp         = (float*)d_out;
    float* out_batch[2] = {outp, outp + (size_t)NB[0] * DIMV};
    float* out_transe0  = outp + ((size_t)NB[0] + NB[1]) * DIMV;
    float* out_transe[2] = {out_transe0, out_transe0 + (size_t)NTRI[0] * DIMV};

    const int TB = 256;
    for (int g = 0; g < 2; ++g) {
        const int N = Ns[g], E = Es[g];
        const int nfeat = N * DIMV;
        for (int l = 0; l < NLAYERSV; ++l) {
            const float* xin = (l == 0) ? emb[g] : xf;
            // weight transpose + f16 convert (small)
            conv_w_kernel<<<(NHEADSV * DIMV * DIMV + TB - 1) / TB, TB, 0, stream>>>(
                Wall + (size_t)l * NHEADSV * DIMV * DIMV, WT);
            // activation f16 convert
            f32_to_f16_kernel<<<(nfeat + TB - 1) / TB, TB, 0, stream>>>(xin, xh, nfeat);
            // WMMA GEMM (TDM-staged A tiles): Hbuf[head] = xh @ W[l,head]
            gemm_wmma_kernel<<<dim3((N + MT - 1) / MT, NHEADSV), TB, 0, stream>>>(
                xh, WT, Hbuf, N);
            // attention logits
            score_kernel<<<(NHEADSV * N + 7) / 8, TB, 0, stream>>>(
                Hbuf, a_src + (size_t)l * NHEADSV * DIMV,
                a_dst + (size_t)l * NHEADSV * DIMV, ssrc, sdst, N);
            // init reductions
            fill_kernel<<<(NHEADSV * N + TB - 1) / TB, TB, 0, stream>>>(mbuf, -3.402823466e38f, NHEADSV * N);
            fill_kernel<<<(NHEADSV * N + TB - 1) / TB, TB, 0, stream>>>(den, 0.f, NHEADSV * N);
            fill_kernel<<<(nfeat + TB - 1) / TB, TB, 0, stream>>>(outacc, 0.f, nfeat);
            // softmax over in-edges (stable) + aggregation
            edge_max_kernel<<<(E + TB - 1) / TB, TB, 0, stream>>>(
                srcs[g], dsts[g], ssrc, sdst, mbuf, E, N);
            edge_exp_kernel<<<(E + TB - 1) / TB, TB, 0, stream>>>(
                srcs[g], dsts[g], ssrc, sdst, mbuf, exbuf, den, E, N);
            edge_aggr_kernel<<<(NHEADSV * E + 3) / 4, TB, 0, stream>>>(
                srcs[g], dsts[g], exbuf, den, Hbuf, outacc, E, N);
            // mean heads + elu
            elu_mean_kernel<<<(nfeat + TB - 1) / TB, TB, 0, stream>>>(outacc, xf, nfeat);
        }
        // outputs for this graph
        gather_kernel<<<((size_t)NB[g] * DIMV + TB - 1) / TB, TB, 0, stream>>>(
            xf, datas[g], out_batch[g], NB[g] * DIMV);
        transe_kernel<<<((size_t)NTRI[g] * DIMV + TB - 1) / TB, TB, 0, stream>>>(
            xf, rel[g], hhs[g], tts[g], rrs[g], out_transe[g], NTRI[g] * DIMV);
    }
}
